// Attention_343597384423
// MI455X (gfx1250) — compile-verified, matrix operand order checked by reading, not run on hardware
//
#include <hip/hip_runtime.h>
#include <stdint.h>

// ---------------------------------------------------------------------------
// Types for CDNA5 WMMA
// ---------------------------------------------------------------------------
typedef __attribute__((ext_vector_type(16))) __bf16 v16bf;
typedef __attribute__((ext_vector_type(2)))  __bf16 v2bf;
typedef __attribute__((ext_vector_type(8)))  float  v8f;

union F8 { v8f v; float f[8]; };
union TileBF { uint32_t u[8]; v16bf v; };

#if defined(__has_builtin)
#if __has_builtin(__builtin_amdgcn_cvt_pk_bf16_f32)
#define HAVE_PK_BF16 1
#endif
#endif

// pack two floats into two bf16 (low = a, high = b); native fptrunc is RNE
__device__ __forceinline__ uint32_t pk2bf(float a, float b) {
#ifdef HAVE_PK_BF16
    auto r = __builtin_amdgcn_cvt_pk_bf16_f32(a, b);
    return __builtin_bit_cast(uint32_t, r);
#else
    v2bf v;
    v[0] = (__bf16)a;
    v[1] = (__bf16)b;
    return __builtin_bit_cast(uint32_t, v);
#endif
}

// Load a 16x32 bf16 tile in the WMMA A/B register layout from a row-major
// [rows][strideElems] bf16 (ushort) buffer (LDS or global; inlined per site).
// ISA 7.12.2 layout: lanes 0-15 hold K quarters 0 (VGPR0-3) and 2 (VGPR4-7),
// lanes 16-31 hold K quarters 1 and 3, row = lane&15.
__device__ __forceinline__ v16bf load_tile(const unsigned short* base,
                                           int strideElems, int row0, int k0,
                                           int lane) {
    int r = row0 + (lane & 15);
    int k = k0 + ((lane >> 4) << 3);      // +8 elements for upper lane half
    const unsigned short* p = base + r * strideElems + k;
    TileBF t;
    uint4 lo = *(const uint4*)(p);        // K quarter 0/1 -> VGPR 0..3
    uint4 hi = *(const uint4*)(p + 16);   // K quarter 2/3 -> VGPR 4..7
    t.u[0] = lo.x; t.u[1] = lo.y; t.u[2] = lo.z; t.u[3] = lo.w;
    t.u[4] = hi.x; t.u[5] = hi.y; t.u[6] = hi.z; t.u[7] = hi.w;
    return t.v;
}

#define WMMA_BF16(a, b, c) \
    __builtin_amdgcn_wmma_f32_16x16x32_bf16(false, (a), false, (b), (short)0, (c), false, false)

// ---------------------------------------------------------------------------
// Problem constants
// ---------------------------------------------------------------------------
#define NTOK   49      // tokens per window (7x7)
#define NP     64      // padded tokens
#define DIMK   128     // model dim
#define HEADS  4
#define DH     32
#define SCALE  0.17677669529663687f  // 32^-0.5

// LDS layout (bytes). Strides padded (+8 bf16 elems) for bank-conflict-free
// ds_load_b128 tile loads.
#define XS 136   // stride (elems) for 128-col bf16 matrices
#define VS 72    // stride for Vt [128][64]
#define AS 72    // stride for attn [64][64] per head
#define OFF_X   0                         // X bf16 [64][XS]  (reused as attn-out)
#define OFF_Q   17408                     // Q bf16 [64][XS] (pre-scaled)
#define OFF_K   34816                     // K bf16 [64][XS]
#define OFF_V   52224                     // Vt bf16 [128][VS] (dim-major)
#define OFF_SIM 70656                     // sim f32 [4][64][64]; aliased attn bf16 [4][64][AS]
#define SMEM_BYTES 136192

// Workspace layout (bytes)
#define WS_WQKVT 0        // bf16 [384][128]  (N-major, i.e. W^T)
#define WS_WOUTT 98304    // bf16 [128][128]
#define WS_BIAS  131072   // f32  [4][64][64]

// ---------------------------------------------------------------------------
// Prep kernel: transpose+convert weights to bf16, build padded bias matrix
// ---------------------------------------------------------------------------
__global__ __launch_bounds__(256) void prep_kernel(
    const float* __restrict__ Wqkv,   // [128][384] (x @ W)
    const float* __restrict__ Wout,   // [128][128]
    const float* __restrict__ btab,   // [(2*7-1)^2 = 169][4]
    unsigned short* __restrict__ wqkvT,
    unsigned short* __restrict__ woutT,
    float* __restrict__ biasx)
{
    int i = blockIdx.x * 256 + threadIdx.x;
    if (i < 384 * 128) {
        int n = i >> 7, k = i & 127;
        wqkvT[n * 128 + k] = (unsigned short)pk2bf(Wqkv[k * 384 + n], 0.0f);
    } else if (i < 384 * 128 + 128 * 128) {
        int j = i - 384 * 128;
        int n = j >> 7, k = j & 127;
        woutT[n * 128 + k] = (unsigned short)pk2bf(Wout[k * 128 + n], 0.0f);
    } else if (i < 384 * 128 + 128 * 128 + HEADS * NP * NP) {
        int j = i - (384 * 128 + 128 * 128);
        int h = j >> 12;
        int rc = j & 4095;
        int r = rc >> 6, c = rc & 63;
        float v = -1e30f;                 // mask padded columns
        if (r < NTOK && c < NTOK) {
            int gi0 = r / 7, gj0 = r % 7, gi1 = c / 7, gj1 = c % 7;
            int idx = (gi0 - gi1 + 6) * 13 + (gj0 - gj1 + 6);
            v = btab[idx * HEADS + h];
        }
        biasx[(h * NP + r) * NP + c] = v;
    }
}

// ---------------------------------------------------------------------------
// Main fused windowed-attention kernel: 1 block (8 waves) per window
// ---------------------------------------------------------------------------
__global__ __launch_bounds__(256) void attn_win_kernel(
    const float* __restrict__ x,              // [8192][49][128]
    const unsigned short* __restrict__ wqkvT, // bf16 [384][128]
    const unsigned short* __restrict__ woutT, // bf16 [128][128]
    const float* __restrict__ biasx,          // [4][64][64]
    float* __restrict__ out)                  // [8192][49][128]
{
    __shared__ __align__(16) unsigned char smem[SMEM_BYTES];
    unsigned short* Xb    = (unsigned short*)(smem + OFF_X);
    unsigned short* Qb    = (unsigned short*)(smem + OFF_Q);
    unsigned short* Kb    = (unsigned short*)(smem + OFF_K);
    unsigned short* Vt    = (unsigned short*)(smem + OFF_V);
    float*          simf  = (float*)(smem + OFF_SIM);
    unsigned short* attnb = (unsigned short*)(smem + OFF_SIM); // alias (barrier-guarded)
    unsigned short* outb  = Xb;                                // alias, X dead by GEMM3

    const int tid  = threadIdx.x;
    const int lane = tid & 31;
    const int wv   = tid >> 5;
    const int cl   = lane & 15;
    const int hi8  = (lane >> 4) << 3;
    const long win = blockIdx.x;
    const float* xw = x + win * (long)(NTOK * DIMK);

    // ---- Stage 0: load X (float4), convert to packed bf16, zero-pad ----
    for (int i = tid; i < NP * (DIMK / 4); i += 256) {   // 2048 float4
        int r = i >> 5, c4 = (i & 31) << 2;
        float4 v = make_float4(0.0f, 0.0f, 0.0f, 0.0f);
        if (r < NTOK) v = *(const float4*)(xw + r * DIMK + c4);
        uint2 p;
        p.x = pk2bf(v.x, v.y);
        p.y = pk2bf(v.z, v.w);
        *(uint2*)(Xb + r * XS + c4) = p;                 // ds_store_b64
    }
    __syncthreads();

    // ---- GEMM1: QKV[64][384] = X[64][128] @ Wqkv ; 96 tiles, 12/wave ----
    for (int j = 0; j < 3; ++j) {
        int nt = wv * 3 + j;                       // 0..23
        v16bf b0 = load_tile(wqkvT, 128, nt * 16, 0,  lane);
        v16bf b1 = load_tile(wqkvT, 128, nt * 16, 32, lane);
        v16bf b2 = load_tile(wqkvT, 128, nt * 16, 64, lane);
        v16bf b3 = load_tile(wqkvT, 128, nt * 16, 96, lane);
        for (int mt = 0; mt < 4; ++mt) {
            F8 c;
            #pragma unroll
            for (int r = 0; r < 8; ++r) c.f[r] = 0.0f;
            v16bf a;
            a = load_tile(Xb, XS, mt * 16, 0,  lane); c.v = WMMA_BF16(a, b0, c.v);
            a = load_tile(Xb, XS, mt * 16, 32, lane); c.v = WMMA_BF16(a, b1, c.v);
            a = load_tile(Xb, XS, mt * 16, 64, lane); c.v = WMMA_BF16(a, b2, c.v);
            a = load_tile(Xb, XS, mt * 16, 96, lane); c.v = WMMA_BF16(a, b3, c.v);
            if (nt < 8) {                      // Q (pre-scaled)
                int cb = nt * 16 + cl;
                #pragma unroll
                for (int r = 0; r < 8; r += 2) {
                    int row = mt * 16 + r + hi8;
                    uint32_t p = pk2bf(c.f[r] * SCALE, c.f[r + 1] * SCALE);
                    Qb[row * XS + cb]       = (unsigned short)p;
                    Qb[(row + 1) * XS + cb] = (unsigned short)(p >> 16);
                }
            } else if (nt < 16) {              // K
                int cb = (nt - 8) * 16 + cl;
                #pragma unroll
                for (int r = 0; r < 8; r += 2) {
                    int row = mt * 16 + r + hi8;
                    uint32_t p = pk2bf(c.f[r], c.f[r + 1]);
                    Kb[row * XS + cb]       = (unsigned short)p;
                    Kb[(row + 1) * XS + cb] = (unsigned short)(p >> 16);
                }
            } else {                           // V, stored transposed [dim][tok]
                int d = (nt - 16) * 16 + cl;
                #pragma unroll
                for (int r = 0; r < 8; r += 2) {
                    int row = mt * 16 + r + hi8;   // even -> b32-aligned pair
                    *(uint32_t*)(Vt + d * VS + row) = pk2bf(c.f[r], c.f[r + 1]);
                }
            }
        }
    }
    __syncthreads();

    // ---- GEMM2: sim[h][64][64] = Qs_h @ K_h^T ; 64 tiles, 8/wave ----
    for (int i = 0; i < 8; ++i) {
        int t = wv * 8 + i;
        int h = t >> 4, rem = t & 15, mt = rem >> 2, nt = rem & 3;
        v16bf a = load_tile(Qb, XS, mt * 16, h * DH, lane);
        v16bf b = load_tile(Kb, XS, nt * 16, h * DH, lane);
        F8 c;
        #pragma unroll
        for (int r = 0; r < 8; ++r) c.f[r] = 0.0f;
        c.v = WMMA_BF16(a, b, c.v);
        int col = nt * 16 + cl;
        #pragma unroll
        for (int r = 0; r < 8; ++r) {
            int row = mt * 16 + r + hi8;
            simf[(h * NP + row) * NP + col] = c.f[r];
        }
    }
    __syncthreads();

    // ---- Softmax (+bias): 256 threads == 4 heads x 64 rows ----
    {
        int h = tid >> 6, row = tid & 63;
        const float4* s4 = (const float4*)(simf  + (h * NP + row) * NP);
        const float4* b4 = (const float4*)(biasx + (h * NP + row) * NP);
        float rbuf[NP];
        float m = -3.4e38f;
        #pragma unroll
        for (int jx = 0; jx < NP / 4; ++jx) {
            float4 sv = s4[jx];
            float4 bv = b4[jx];
            float v0 = sv.x + bv.x, v1 = sv.y + bv.y;
            float v2 = sv.z + bv.z, v3 = sv.w + bv.w;
            rbuf[4 * jx + 0] = v0; rbuf[4 * jx + 1] = v1;
            rbuf[4 * jx + 2] = v2; rbuf[4 * jx + 3] = v3;
            m = fmaxf(m, fmaxf(fmaxf(v0, v1), fmaxf(v2, v3)));
        }
        float s = 0.0f;
        #pragma unroll
        for (int jx = 0; jx < NP; ++jx) { float e = __expf(rbuf[jx] - m); rbuf[jx] = e; s += e; }
        float inv = 1.0f / s;
        __syncthreads();   // all reads of simf done before bf16 overwrite
        uint32_t* arow = (uint32_t*)(attnb + (h * NP + row) * AS);
        #pragma unroll
        for (int jx = 0; jx < NP / 2; ++jx)
            arow[jx] = pk2bf(rbuf[2 * jx] * inv, rbuf[2 * jx + 1] * inv);
    }
    __syncthreads();

    // ---- GEMM3: out_h[64][32] = attn_h[64][64] @ V_h ; 32 tiles, 4/wave ----
    for (int i = 0; i < 4; ++i) {
        int t = wv * 4 + i;
        int h = t >> 3, rem = t & 7, mt = rem >> 1, nt = rem & 1;
        F8 c;
        #pragma unroll
        for (int r = 0; r < 8; ++r) c.f[r] = 0.0f;
        for (int kk = 0; kk < 2; ++kk) {
            v16bf a = load_tile(attnb + h * NP * AS, AS, mt * 16, kk * 32, lane);
            v16bf b = load_tile(Vt + (h * DH) * VS, VS, nt * 16, kk * 32, lane);
            c.v = WMMA_BF16(a, b, c.v);
        }
        int cb = h * DH + nt * 16 + cl;
        #pragma unroll
        for (int r = 0; r < 8; r += 2) {
            int row = mt * 16 + r + hi8;
            uint32_t p = pk2bf(c.f[r], c.f[r + 1]);
            outb[row * XS + cb]       = (unsigned short)p;
            outb[(row + 1) * XS + cb] = (unsigned short)(p >> 16);
        }
    }
    __syncthreads();

    // ---- GEMM4: final[64][128] = attnout @ Wout ; 32 tiles, 4/wave ----
    float* og = out + win * (long)(NTOK * DIMK);
    for (int i = 0; i < 4; ++i) {
        int t = wv * 4 + i;
        int mt = t >> 3, nt = t & 7;
        F8 c;
        #pragma unroll
        for (int r = 0; r < 8; ++r) c.f[r] = 0.0f;
        for (int kk = 0; kk < 4; ++kk) {
            v16bf a = load_tile(outb, XS, mt * 16, kk * 32, lane);
            v16bf b = load_tile(woutT, 128, nt * 16, kk * 32, lane);
            c.v = WMMA_BF16(a, b, c.v);
        }
        int col = nt * 16 + cl;
        #pragma unroll
        for (int r = 0; r < 8; ++r) {
            int row = mt * 16 + r + hi8;
            if (row < NTOK) og[row * DIMK + col] = c.f[r];
        }
    }
}

// ---------------------------------------------------------------------------
// Launch
// ---------------------------------------------------------------------------
extern "C" void kernel_launch(void* const* d_in, const int* in_sizes, int n_in,
                              void* d_out, int out_size, void* d_ws, size_t ws_size,
                              hipStream_t stream) {
    (void)in_sizes; (void)n_in; (void)out_size; (void)ws_size;
    const float* x    = (const float*)d_in[0];
    const float* Wqkv = (const float*)d_in[1];
    const float* Wout = (const float*)d_in[2];
    const float* btab = (const float*)d_in[3];

    unsigned short* wqkvT = (unsigned short*)((char*)d_ws + WS_WQKVT);
    unsigned short* woutT = (unsigned short*)((char*)d_ws + WS_WOUTT);
    float*          biasx = (float*)((char*)d_ws + WS_BIAS);

    // 384*128 + 128*128 + 4*64*64 = 81920 work items
    prep_kernel<<<320, 256, 0, stream>>>(Wqkv, Wout, btab, wqkvT, woutT, biasx);

    attn_win_kernel<<<8192, 256, 0, stream>>>(x, wqkvT, woutT, biasx, (float*)d_out);
}
